// OTHead_6562710028451
// MI455X (gfx1250) — matrix operand order repaired and checked by reading, not compiled, not run on hardware
//
#include <hip/hip_runtime.h>
#include <hip/hip_bf16.h>

// Problem dims (fixed by the reference)
#define NROWS 32768
#define DDIM  512
#define KCOLS 2048
#define WAVES_PER_BLOCK 8        // 8 waves * 16 rows = 128 rows per block
#define CTILES (KCOLS / 16)      // 128 column tiles

typedef __attribute__((ext_vector_type(16))) __bf16 v16bf;
typedef __attribute__((ext_vector_type(8)))  __bf16 v8bf;
typedef __attribute__((ext_vector_type(4)))  __bf16 v4bf;
typedef __attribute__((ext_vector_type(8)))  float  v8f;
typedef __attribute__((ext_vector_type(4)))  unsigned int u32x4;
typedef __attribute__((ext_vector_type(8)))  int i32x8;
typedef __attribute__((ext_vector_type(4)))  int i32x4;

// ---------------------------------------------------------------------------
// Kernel 1: Pn = P / max(sum|P|, eps)  (row-wise L1 norm), stored as bf16.
// ---------------------------------------------------------------------------
__global__ void l1norm_pn(const float* __restrict__ P, __bf16* __restrict__ Pn) {
    const int row = blockIdx.x;
    const int t = threadIdx.x;
    __shared__ float red[256];

    float s = 0.0f;
    for (int i = t; i < DDIM; i += 256)
        s += fabsf(P[(size_t)row * DDIM + i]);
    red[t] = s;
    __syncthreads();
    for (int off = 128; off > 0; off >>= 1) {
        if (t < off) red[t] += red[t + off];
        __syncthreads();
    }
    const float inv = 1.0f / fmaxf(red[0], 1e-12f);
    for (int i = t; i < DDIM; i += 256)
        Pn[(size_t)row * DDIM + i] = (__bf16)(P[(size_t)row * DDIM + i] * inv);
}

// ---------------------------------------------------------------------------
// TDM: DMA one 16-row x 512-col bf16 tile of Pn into LDS (one D#, 2D tensor).
// ---------------------------------------------------------------------------
__device__ __forceinline__ void tdm_load_btile(const __bf16* gsrc, unsigned lds_byte_addr) {
    const unsigned long long ga = (unsigned long long)(size_t)(const void*)gsrc;
    u32x4 g0;
    g0[0] = 1u;                                       // count=1, user descriptor
    g0[1] = lds_byte_addr;                            // lds_addr [63:32]
    g0[2] = (unsigned)(ga & 0xffffffffu);             // global_addr [95:64]
    g0[3] = (unsigned)((ga >> 32) & 0x01ffffffu)      // global_addr [120:96]
          | (2u << 30);                               // type=2 ("image") [127:126]
    i32x8 g1;
    g1[0] = (int)(1u << 16);        // workgroup_mask=0, data_size=1 (2 bytes)
    g1[1] = (int)(512u << 16);      // tensor_dim0 = 512   (bits 79:48, low half)
    g1[2] = (int)(2048u << 16);     // tensor_dim1 = 2048  (bits 111:80, low half)
    g1[3] = (int)(512u << 16);      // tile_dim0 = 512     (bits 127:112)
    g1[4] = 16;                     // tile_dim1 = 16      (bits 143:128)
    g1[5] = 512;                    // tensor_dim0_stride = 512 (bits 207:160)
    g1[6] = 0;
    g1[7] = 0;
    const i32x4 z4 = {0, 0, 0, 0};
#if defined(__clang_major__) && (__clang_major__ >= 23)
    const i32x8 z8 = {0, 0, 0, 0, 0, 0, 0, 0};
    __builtin_amdgcn_tensor_load_to_lds(g0, g1, z4, z4, z8, 0);
#else
    __builtin_amdgcn_tensor_load_to_lds(g0, g1, z4, z4, 0);
#endif
}

// ---------------------------------------------------------------------------
// Kernel 2: fused scr = (z @ Pn.T)*2, online softmax stats + argmax per row.
// Each wave: 16 rows x all 2048 columns via v_wmma_f32_16x16x32_bf16.
// B tiles are TDM-DMA'd into LDS once per block, double-buffered.
// ---------------------------------------------------------------------------
__global__ void score_softmax_argmax(const float* __restrict__ z,
                                     const __bf16* __restrict__ Pn,
                                     float* __restrict__ otRow,
                                     int* __restrict__ idxRow) {
    // LDS layout: [A staging: 8 waves * 16*512 bf16 = 128 KB][B: 2 * 16*512 bf16 = 32 KB]
    extern __shared__ __bf16 zs[];

    const int lane   = threadIdx.x & 31;
    const int wave   = threadIdx.x >> 5;
    const int laneHi = lane >> 4;    // which 16-lane half
    const int lp     = lane & 15;
    const int rowBase = (blockIdx.x * WAVES_PER_BLOCK + wave) * 16;

    __bf16* zw = zs + (size_t)wave * 16 * DDIM;
    __bf16* bbase = zs + (size_t)WAVES_PER_BLOCK * 16 * DDIM;
    const unsigned bOff0 = (unsigned)(size_t)(void*)bbase;   // LDS byte address of B buf 0

    // Stage this wave's 16 z rows into LDS as bf16 (rows are contiguous).
    {
        const float4* src = (const float4*)(z + (size_t)rowBase * DDIM);
        v4bf* dst = (v4bf*)zw;
        for (int i = lane; i < 16 * DDIM / 4; i += 32) {
            const float4 f = src[i];
            v4bf h;
            h.x = (__bf16)f.x; h.y = (__bf16)f.y;
            h.z = (__bf16)f.z; h.w = (__bf16)f.w;
            dst[i] = h;
        }
    }

    // Kick off DMA of B tile 0 into buffer 0.
    if (wave == 0)
        tdm_load_btile(Pn, bOff0);
    __syncthreads();

    // Per-lane online stats for 8 rows (row = rowBase + laneHi*8 + r,
    // columns c with c%16 == lp).
    float m[8], Ae[8], Be[8], Ce[8];
    int bidx[8];
#pragma unroll
    for (int r = 0; r < 8; ++r) {
        m[r] = -__builtin_inff();
        Ae[r] = 0.0f; Be[r] = 0.0f; Ce[r] = 0.0f; bidx[r] = 0;
    }

    for (int ct = 0; ct < CTILES; ++ct) {
        if (wave == 0)
            __builtin_amdgcn_s_wait_tensorcnt(0);   // tile ct landed in LDS
        __syncthreads();  // publish tile ct; everyone done reading tile ct-1
        if (wave == 0 && (ct + 1) < CTILES)
            tdm_load_btile(Pn + (size_t)(ct + 1) * 16 * DDIM,
                           bOff0 + (unsigned)(((ct + 1) & 1) * 16 * DDIM * 2));

        const __bf16* bt = bbase + (size_t)(ct & 1) * 16 * DDIM;
        const __bf16* bp = bt + (size_t)lp * DDIM;   // this lane's Pn row (column of scr)
        v8f acc = {};

#pragma unroll
        for (int kk = 0; kk < DDIM / 32; ++kk) {
            const int koff = kk * 32;
            // A fragment: lane holds row M=lp, K-chunks {koff+hi*8..+8} and {+16..}
            const v8bf a0 = *(const v8bf*)(zw + (size_t)lp * DDIM + koff + laneHi * 8);
            const v8bf a1 = *(const v8bf*)(zw + (size_t)lp * DDIM + koff + 16 + laneHi * 8);
            const v16bf a = __builtin_shufflevector(a0, a1,
                0, 1, 2, 3, 4, 5, 6, 7, 8, 9, 10, 11, 12, 13, 14, 15);
            // B fragment: lane holds column N=lp, K-chunks {koff+hi*8..+8} and {+16..}
            const v8bf b0 = *(const v8bf*)(bp + koff + laneHi * 8);
            const v8bf b1 = *(const v8bf*)(bp + koff + 16 + laneHi * 8);
            const v16bf b = __builtin_shufflevector(b0, b1,
                0, 1, 2, 3, 4, 5, 6, 7, 8, 9, 10, 11, 12, 13, 14, 15);
            acc = __builtin_amdgcn_wmma_f32_16x16x32_bf16(
                false, a, false, b, (short)0, acc, false, false);
        }

        const int col = ct * 16 + lp;
#pragma unroll
        for (int r = 0; r < 8; ++r) {
            const float v = acc[r] * 2.0f;  // scr = dot / TEMP_SCORE
            if (v > m[r]) {
                const float d1 = m[r] - v;           // <= 0 (or -inf first time)
                const float e1 = __expf(d1);
                const float e10 = __expf(d1 * 10.0f);
                Ae[r] = Ae[r] * e1 + 1.0f;
                Be[r] = Be[r] * e10 + 1.0f;
                Ce[r] = Ce[r] * e10 + v;
                m[r] = v;
                bidx[r] = col;
            } else {
                const float d = v - m[r];
                const float e1 = __expf(d);
                const float e10 = __expf(d * 10.0f);
                Ae[r] += e1;
                Be[r] += e10;
                Ce[r] += v * e10;
            }
        }
    }

    // Merge stats across the 16 lanes sharing each row (xor 1..8 stays in half).
#pragma unroll
    for (int r = 0; r < 8; ++r) {
        float mr = m[r], Ar = Ae[r], Br = Be[r], Cr = Ce[r];
        int ir = bidx[r];
#pragma unroll
        for (int off = 1; off < 16; off <<= 1) {
            const float mo = __shfl_xor(mr, off, 32);
            const float Ao = __shfl_xor(Ar, off, 32);
            const float Bo = __shfl_xor(Br, off, 32);
            const float Co = __shfl_xor(Cr, off, 32);
            const int   io = __shfl_xor(ir, off, 32);
            const float mn = fmaxf(mr, mo);
            const float s1a  = __expf(mr - mn);
            const float s1b  = __expf(mo - mn);
            const float s10a = __expf((mr - mn) * 10.0f);
            const float s10b = __expf((mo - mn) * 10.0f);
            Ar = Ar * s1a + Ao * s1b;
            Br = Br * s10a + Bo * s10b;
            Cr = Cr * s10a + Co * s10b;
            ir = (mo > mr || (mo == mr && io < ir)) ? io : ir;  // first-max tiebreak
            mr = mn;
        }
        if (lp == 0) {
            const int row = rowBase + laneHi * 8 + r;
            // ot_row = logsumexp(scr) - sum_k Q_k * scr_k
            otRow[row] = mr + __logf(Ar) - Cr / Br;
            idxRow[row] = ir;
        }
    }
}

// ---------------------------------------------------------------------------
// Kernel 3: ot = mean(otRow) -> d_out[0]
// ---------------------------------------------------------------------------
__global__ void reduce_ot(const float* __restrict__ otRow, float* __restrict__ out) {
    __shared__ float red[256];
    const int t = threadIdx.x;
    float s = 0.0f;
    for (int i = t; i < NROWS; i += 256) s += otRow[i];
    red[t] = s;
    __syncthreads();
    for (int off = 128; off > 0; off >>= 1) {
        if (t < off) red[t] += red[t + off];
        __syncthreads();
    }
    if (t == 0) out[0] = red[0] / (float)NROWS;
}

// ---------------------------------------------------------------------------
// Kernel 4: cent[row] = P[idx[row]] ; dest is only 4B-aligned (out+1),
// so read float4, store scalar floats.
// ---------------------------------------------------------------------------
__global__ void gather_cent(const float* __restrict__ P,
                            const int* __restrict__ idxRow,
                            float* __restrict__ cent) {
    const int row = blockIdx.x;
    const int t = threadIdx.x;  // 128 threads, 4 floats each
    const int k = idxRow[row];
    const float4 v = ((const float4*)(P + (size_t)k * DDIM))[t];
    float* dst = cent + (size_t)row * DDIM + (size_t)t * 4;
    dst[0] = v.x; dst[1] = v.y; dst[2] = v.z; dst[3] = v.w;
}

// ---------------------------------------------------------------------------
extern "C" void kernel_launch(void* const* d_in, const int* in_sizes, int n_in,
                              void* d_out, int out_size, void* d_ws, size_t ws_size,
                              hipStream_t stream) {
    const float* z = (const float*)d_in[0];   // [N, d] fp32
    const float* P = (const float*)d_in[1];   // [K, d] fp32
    float* out = (float*)d_out;               // [1 + N*d] fp32: ot, cent

    // Workspace layout: Pn bf16 (K*d*2 B) | otRow (N*4 B) | idxRow (N*4 B)
    __bf16* pn   = (__bf16*)d_ws;
    float* otRow = (float*)((char*)d_ws + (size_t)KCOLS * DDIM * 2);
    int* idxRow  = (int*)((char*)d_ws + (size_t)KCOLS * DDIM * 2 + (size_t)NROWS * 4);

    l1norm_pn<<<KCOLS, 256, 0, stream>>>(P, pn);

    const int blocks = NROWS / (16 * WAVES_PER_BLOCK);  // 256
    const size_t smem = (size_t)WAVES_PER_BLOCK * 16 * DDIM * 2   // A staging 128 KB
                      + (size_t)2 * 16 * DDIM * 2;                // B double buffer 32 KB
    score_softmax_argmax<<<blocks, 32 * WAVES_PER_BLOCK, smem, stream>>>(
        z, pn, otRow, idxRow);

    reduce_ot<<<1, 256, 0, stream>>>(otRow, out);

    gather_cent<<<NROWS, 128, 0, stream>>>(P, idxRow, out + 1);
}
